// mLSTMLayer_42915313222193
// MI455X (gfx1250) — matrix operand order repaired and checked by reading, not compiled, not run on hardware
//
#include <hip/hip_runtime.h>
#include <hip/hip_bf16.h>
#include <math.h>

typedef __attribute__((ext_vector_type(16))) _Float16 v16h;
typedef __attribute__((ext_vector_type(8)))  float    v8f;
typedef __attribute__((__vector_size__(16))) int      v4i_;

#define EMB     1024
#define NH      4
#define INNER   2048
#define DH      512
#define KS      4
#define SEQ     2048
#define BDIM    4
#define XLD     4096            // x_inner leading dim (2*INNER)
#define BS_TOT  (BDIM * SEQ)    // 8192

// ---------------------------------------------------------------------------
// gfx1250 async global->LDS copy (ASYNCcnt path), with compile-safe fallback.
// Builtin proto: (V4i addrspace(1)*, V4i addrspace(3)*, imm offset, imm cpol)
// ---------------------------------------------------------------------------
#if __has_builtin(__builtin_amdgcn_global_load_async_to_lds_b128)
#define HAS_ASYNC_LDS 1
#else
#define HAS_ASYNC_LDS 0
#endif

__device__ __forceinline__ void async_g2l_b128(const void* g, void* l) {
#if HAS_ASYNC_LDS
  __builtin_amdgcn_global_load_async_to_lds_b128(
      (__attribute__((address_space(1))) v4i_*)g,
      (__attribute__((address_space(3))) v4i_*)l, 0, 0);
#else
  *(v4i_*)l = *(const v4i_*)g;
#endif
}

__device__ __forceinline__ void wait_async_lds() {
#if HAS_ASYNC_LDS
#if __has_builtin(__builtin_amdgcn_s_wait_asynccnt)
  __builtin_amdgcn_s_wait_asynccnt(0);
#else
  asm volatile("s_wait_asynccnt 0x0" ::: "memory");
#endif
#endif
}

// ---------------------------------------------------------------------------
// WMMA helpers (CDNA5 16x16x32 f16 -> f32)
// ---------------------------------------------------------------------------
__device__ __forceinline__ v8f wmma16(v16h a, v16h b, v8f c) {
  // (neg_a, A, neg_b, B, c_mod, C, reuse_a, reuse_b)
  return __builtin_amdgcn_wmma_f32_16x16x32_f16(false, a, false, b, (short)0, c,
                                                false, false);
}

// A operand fragment from row-major [M x K] f16 tile.
// Layout (contiguous, matching B): lanes 0-15 -> K 0..15,
// lanes 16-31 -> K 16..31.
__device__ __forceinline__ v16h frag_a_h(const _Float16* t, int ld, int m0,
                                         int k0, int lane) {
  const _Float16* p = t + (size_t)(m0 + (lane & 15)) * ld + k0;
  const int lh = (lane & 16);  // 0 or 16
  v16h r;
#pragma unroll
  for (int i = 0; i < 16; ++i) r[i] = p[lh + i];
  return r;
}

// B operand fragment from row-major [N x K] storage (column n == row n of
// storage). Contiguous k order, same as A: lanes 0-15 -> K 0..15, lanes 16-31 -> 16..31.
__device__ __forceinline__ v16h frag_b_nk_h(const _Float16* t, int ld, int n0,
                                            int k0, int lane) {
  const _Float16* p = t + (size_t)(n0 + (lane & 15)) * ld + k0;
  const int kh = (lane & 16);
  v16h r;
#pragma unroll
  for (int i = 0; i < 16; ++i) r[i] = p[kh + i];
  return r;
}

// B operand fragment from row-major [K x N] f16 storage.
__device__ __forceinline__ v16h frag_b_kn_h(const _Float16* t, int ld, int k0,
                                            int n0, int lane) {
  const _Float16* p = t + n0 + (lane & 15);
  const int kh = (lane & 16);
  v16h r;
#pragma unroll
  for (int i = 0; i < 16; ++i) r[i] = p[(size_t)(k0 + kh + i) * ld];
  return r;
}

// Pack 4 f32 -> 4 f16 and store as one b64 write.
__device__ __forceinline__ void store_h4(_Float16* dst, float4 v, float s) {
  union {
    _Float16 h[4];
    unsigned long long u;
  } p;
  p.h[0] = (_Float16)(v.x * s);
  p.h[1] = (_Float16)(v.y * s);
  p.h[2] = (_Float16)(v.z * s);
  p.h[3] = (_Float16)(v.w * s);
  *(unsigned long long*)dst = p.u;
}

// ---------------------------------------------------------------------------
// Bulk f32 -> f16 conversion (one-time, for x / w_up / w_down).
// ---------------------------------------------------------------------------
__global__ __launch_bounds__(256) void cvt_f32_f16_kernel(
    const float* __restrict__ src, _Float16* __restrict__ dst) {
  const size_t vi = (size_t)blockIdx.x * 256 + threadIdx.x;
  const float4 v = *(const float4*)&src[vi * 4];
  store_h4(&dst[vi * 4], v, 1.f);
}

// ---------------------------------------------------------------------------
// f16 GEMM: C[M,N] = sum_k A[m,k] * B[n,k]   (A: MxK f16, B: NxK f16)
// 128x128 block tile, BK=32, 8 waves in a 4(M)x2(N) grid, 2x4 frags per wave.
// Tiles staged raw f16 via async global->LDS b128; inner loop is pure
// ds_load_b128 + v_wmma (no conversions anywhere).
// ---------------------------------------------------------------------------
__global__ __launch_bounds__(256) void gemm_h_bt_wmma(
    const _Float16* __restrict__ A, const _Float16* __restrict__ B,
    float* __restrict__ C, int M, int N, int K, int lda, int ldb, int ldc) {
  __shared__ _Float16 As[128 * 32];
  __shared__ _Float16 Bs[128 * 32];
  const int tid = threadIdx.x, lane = tid & 31, wave = tid >> 5;
  const int wm = wave & 3, wn = wave >> 2;
  const int m0 = blockIdx.y * 128, n0 = blockIdx.x * 128;
  const int wm0 = wm * 32, wn0 = wn * 64;
  v8f acc[2][4] = {};
  for (int kk = 0; kk < K; kk += 32) {
    __syncthreads();  // previous tile's fragment reads complete
#pragma unroll
    for (int i = 0; i < 2; ++i) {
      const int vi = tid + i * 256;  // 512 b128 chunks per 128x32 f16 tile
      const int r = vi >> 2, c8 = (vi & 3) * 8;
      async_g2l_b128(&A[(size_t)(m0 + r) * lda + kk + c8], &As[r * 32 + c8]);
      async_g2l_b128(&B[(size_t)(n0 + r) * ldb + kk + c8], &Bs[r * 32 + c8]);
    }
    if (kk + 32 < K) {  // speculative prefetch of the next K-slab
      __builtin_prefetch(&A[(size_t)(m0 + (tid >> 1)) * lda + kk + 32], 0, 0);
      __builtin_prefetch(&B[(size_t)(n0 + (tid >> 1)) * ldb + kk + 32], 0, 0);
    }
    wait_async_lds();
    __syncthreads();
    const v16h af0 = frag_a_h(As, 32, wm0, 0, lane);
    const v16h af1 = frag_a_h(As, 32, wm0 + 16, 0, lane);
#pragma unroll
    for (int j = 0; j < 4; ++j) {
      const v16h bf = frag_b_nk_h(Bs, 32, wn0 + 16 * j, 0, lane);
      acc[0][j] = wmma16(af0, bf, acc[0][j]);
      acc[1][j] = wmma16(af1, bf, acc[1][j]);
    }
  }
  const int ch = (lane & 16) >> 1;  // +8 rows for hi half-wave
#pragma unroll
  for (int i2 = 0; i2 < 2; ++i2)
#pragma unroll
    for (int j = 0; j < 4; ++j) {
      const int row0 = m0 + wm0 + 16 * i2 + ch;
      const int col = n0 + wn0 + 16 * j + (lane & 15);
#pragma unroll
      for (int r = 0; r < 8; ++r)
        C[(size_t)(row0 + r) * ldc + col] = acc[i2][j][r];
    }
}

// ---------------------------------------------------------------------------
// Causal depthwise conv1d (K=4) + SiLU on the mLSTM branch of x_inner.
// ---------------------------------------------------------------------------
__global__ __launch_bounds__(256) void conv_silu_kernel(
    const float* __restrict__ x_inner, const float* __restrict__ cw,
    const float* __restrict__ cb, float* __restrict__ xc) {
  const int idx = blockIdx.x * 256 + threadIdx.x;  // over B*S*INNER
  const int c = idx & (INNER - 1);
  const int bs = idx >> 11;
  const int b = bs >> 11, s = bs & (SEQ - 1);
  float acc = cb[c];
#pragma unroll
  for (int j = 0; j < KS; ++j) {
    const int sp = s - (KS - 1) + j;
    if (sp >= 0)
      acc += x_inner[(size_t)(b * SEQ + sp) * XLD + c] * cw[c * KS + j];
  }
  xc[idx] = acc / (1.f + __expf(-acc));  // SiLU
}

// ---------------------------------------------------------------------------
// LinearHeadwiseExpand: block-diagonal 4x4 projections; q,k from conv branch,
// v from raw branch. Outputs written directly in f16 for the WMMA consumers.
// ---------------------------------------------------------------------------
__global__ __launch_bounds__(256) void qkv_kernel(
    const float* __restrict__ xc, const float* __restrict__ x_inner,
    const float* __restrict__ wq, const float* __restrict__ wk,
    const float* __restrict__ wv, _Float16* __restrict__ q,
    _Float16* __restrict__ k, _Float16* __restrict__ v) {
  const int idx = blockIdx.x * 256 + threadIdx.x;
  const int c = idx & (INNER - 1);
  const int bs = idx >> 11;
  const int hp = c >> 2, o = c & 3;
  float aq = 0.f, ak = 0.f, av = 0.f;
#pragma unroll
  for (int d = 0; d < 4; ++d) {
    const float xcv = xc[(size_t)bs * INNER + hp * 4 + d];
    const float xmv = x_inner[(size_t)bs * XLD + hp * 4 + d];
    const int w = hp * 16 + o * 4 + d;
    aq += xcv * wq[w];
    ak += xcv * wk[w];
    av += xmv * wv[w];
  }
  q[idx] = (_Float16)aq;
  k[idx] = (_Float16)ak;
  v[idx] = (_Float16)av;
}

// ---------------------------------------------------------------------------
// Gate pre-activations: igp/fgp[b,h,s] = concat(q,k,v).w_*[h] + b_*[h]
// (f16 inputs, f32 accumulation)
// ---------------------------------------------------------------------------
__global__ __launch_bounds__(256) void gate_kernel(
    const _Float16* __restrict__ q, const _Float16* __restrict__ k,
    const _Float16* __restrict__ v, const float* __restrict__ wig,
    const float* __restrict__ big, const float* __restrict__ wfg,
    const float* __restrict__ bfg, float* __restrict__ igp,
    float* __restrict__ fgp) {
  const int bs = blockIdx.x, h = blockIdx.y, tid = threadIdx.x;
  const float* wi = wig + (size_t)h * 3 * INNER;
  const float* wf = wfg + (size_t)h * 3 * INNER;
  float ai = 0.f, af = 0.f;
  for (int c = tid; c < INNER; c += 256) {
    const float qv = (float)q[(size_t)bs * INNER + c];
    const float kv = (float)k[(size_t)bs * INNER + c];
    const float vv = (float)v[(size_t)bs * INNER + c];
    ai += qv * wi[c] + kv * wi[INNER + c] + vv * wi[2 * INNER + c];
    af += qv * wf[c] + kv * wf[INNER + c] + vv * wf[2 * INNER + c];
  }
  __shared__ float r1[256];
  __shared__ float r2[256];
  r1[tid] = ai;
  r2[tid] = af;
  __syncthreads();
  for (int o = 128; o > 0; o >>= 1) {
    if (tid < o) {
      r1[tid] += r1[tid + o];
      r2[tid] += r2[tid + o];
    }
    __syncthreads();
  }
  if (tid == 0) {
    const int b = bs >> 11, s = bs & (SEQ - 1);
    const size_t off = ((size_t)b * NH + h) * SEQ + s;
    igp[off] = r1[0] + big[h];
    fgp[off] = r2[0] + bfg[h];
  }
}

// ---------------------------------------------------------------------------
// Per-(b,h) scan: a[s] = cumsum(logsigmoid(fgp)), g[s] = igp[s]-a[s],
// m[s] = a[s] + prefixmax_{t<=s} g[t]  (exact rowwise stabilizer of log_D).
// ---------------------------------------------------------------------------
__global__ __launch_bounds__(256) void scan_kernel(
    const float* __restrict__ fgp, const float* __restrict__ igp,
    float* __restrict__ a_out, float* __restrict__ g_out,
    float* __restrict__ m_out) {
  const int bh = blockIdx.x, tid = threadIdx.x;
  const size_t base = (size_t)bh * SEQ;
  const int t0 = tid * 8;
  float av[8];
  float run = 0.f;
#pragma unroll
  for (int j = 0; j < 8; ++j) {
    const float x = fgp[base + t0 + j];
    const float lf = fminf(x, 0.f) - log1pf(__expf(-fabsf(x)));  // logsigmoid
    run += lf;
    av[j] = run;
  }
  __shared__ float sb[256];
  sb[tid] = run;
  __syncthreads();
  for (int off = 1; off < 256; off <<= 1) {
    const float p = (tid >= off) ? sb[tid - off] : 0.f;
    __syncthreads();
    sb[tid] += p;
    __syncthreads();
  }
  const float excl = sb[tid] - run;
  float aj[8], gm[8];
  float runm = -3.4e38f;
#pragma unroll
  for (int j = 0; j < 8; ++j) {
    aj[j] = excl + av[j];
    a_out[base + t0 + j] = aj[j];
    const float g = igp[base + t0 + j] - aj[j];
    g_out[base + t0 + j] = g;
    runm = fmaxf(runm, g);
    gm[j] = runm;
  }
  __syncthreads();
  sb[tid] = runm;
  __syncthreads();
  for (int off = 1; off < 256; off <<= 1) {
    const float p = (tid >= off) ? sb[tid - off] : -3.4e38f;
    __syncthreads();
    sb[tid] = fmaxf(sb[tid], p);
    __syncthreads();
  }
  const float em = (tid > 0) ? sb[tid - 1] : -3.4e38f;
#pragma unroll
  for (int j = 0; j < 8; ++j)
    m_out[base + t0 + j] = aj[j] + fmaxf(em, gm[j]);
}

// ---------------------------------------------------------------------------
// Quadratic mLSTM core, flash-style per 64-row query tile; no online rescale
// because m[s] is the exact row max of log_D. The 1/sqrt(DH) scale is folded
// into the gating exponent, so Q/K/V tiles stage as raw f16 bytes via async
// global->LDS DMA.
//   P[s,t] = qk * exp(a[s]-m[s]+g[t]+ln(1/sqrt(DH))),  t <= s
//   h[s,:] = (P @ V) / (max(|sum_t P|, exp(-m[s])) + 1e-6)
// 8 waves = 4(M-strip of 16 rows) x 2(DH halves of 256 cols).
// ---------------------------------------------------------------------------
__global__ __launch_bounds__(256) void mlstm_attn_kernel(
    const _Float16* __restrict__ q, const _Float16* __restrict__ kmat,
    const _Float16* __restrict__ vmat, const float* __restrict__ a_cum,
    const float* __restrict__ g_vec, const float* __restrict__ m_row,
    float* __restrict__ h_out) {
  __shared__ _Float16 Qs[64 * 512];  // query tile
  __shared__ _Float16 Ks[64 * 512];  // key tile (keys x DH)
  __shared__ _Float16 Vc[64 * 512];  // value tile (keys x DH)
  __shared__ float Sf[64 * 64];      // raw scores
  __shared__ _Float16 Pf[64 * 64];   // gated probabilities (f16)
  __shared__ float rowE[64];         // a[s]-m[s]+ln(invs), later 1/norm
  __shared__ float gt[64];
  __shared__ float rowsum[64];

  const int tid = threadIdx.x, lane = tid & 31, wave = tid >> 5;
  const int mw = wave & 3, nw = wave >> 2;
  const int qt = blockIdx.x, h = blockIdx.y, b = blockIdx.z;
  const int sq0 = qt * 64;
  const float log_invs = -3.1191623f;  // ln(1/sqrt(512))

#pragma unroll 4
  for (int i = 0; i < 16; ++i) {  // 4096 b128 chunks per 64x512 f16 tile
    const int vi = tid + i * 256;
    const int r = vi >> 6, c8 = (vi & 63) * 8;
    async_g2l_b128(&q[(size_t)(b * SEQ + sq0 + r) * INNER + h * DH + c8],
                   &Qs[r * 512 + c8]);
  }
  if (tid < 64) {
    const size_t off = ((size_t)b * NH + h) * SEQ + sq0 + tid;
    rowE[tid] = a_cum[off] - m_row[off] + log_invs;
    rowsum[tid] = 0.f;
  }
  const v8f vzero = {};
  v8f acc[16];
#pragma unroll
  for (int t = 0; t < 16; ++t) acc[t] = vzero;

  for (int kt = 0; kt <= qt; ++kt) {
    const int st0 = kt * 64;
    __syncthreads();  // previous iteration's K/V/Pf reads are done
    if (tid < 64) gt[tid] = g_vec[((size_t)b * NH + h) * SEQ + st0 + tid];
#pragma unroll 4
    for (int i = 0; i < 16; ++i) {
      const int vi = tid + i * 256;
      const int r = vi >> 6, c8 = (vi & 63) * 8;
      const size_t goff = (size_t)(b * SEQ + st0 + r) * INNER + h * DH + c8;
      async_g2l_b128(&kmat[goff], &Ks[r * 512 + c8]);
      async_g2l_b128(&vmat[goff], &Vc[r * 512 + c8]);
    }
    wait_async_lds();  // also covers the Q tile on the first iteration
    __syncthreads();
    // ---- S = Q . K^T, all operands LDS-resident ---------------------------
    v8f s0 = vzero, s1 = vzero;
#pragma unroll 4
    for (int kc = 0; kc < 16; ++kc) {
      const v16h aQ = frag_a_h(Qs, 512, mw * 16, kc * 32, lane);
      const v16h b0 = frag_b_nk_h(Ks, 512, (2 * nw) * 16, kc * 32, lane);
      const v16h b1 = frag_b_nk_h(Ks, 512, (2 * nw + 1) * 16, kc * 32, lane);
      s0 = wmma16(aQ, b0, s0);
      s1 = wmma16(aQ, b1, s1);
    }
    {
      const int ch = (lane & 16) >> 1;
      const int row0 = mw * 16 + ch, col = lane & 15;
#pragma unroll
      for (int r = 0; r < 8; ++r) {
        Sf[(row0 + r) * 64 + (2 * nw) * 16 + col] = s0[r];
        Sf[(row0 + r) * 64 + (2 * nw + 1) * 16 + col] = s1[r];
      }
    }
    __syncthreads();
    // ---- gating: P = S * exp(a-m+g+ln(invs)), causal mask on diagonal -----
    const bool diag = (kt == qt);
    for (int e = tid; e < 64 * 64; e += 256) {
      const int i = e >> 6, j = e & 63;
      float pv = 0.f;
      if (!diag || j <= i) pv = Sf[e] * __expf(rowE[i] + gt[j]);
      Pf[e] = (_Float16)pv;
      atomicAdd(&rowsum[i], pv);  // ds_add_f32
    }
    __syncthreads();
    // ---- acc += P . V ------------------------------------------------------
    const v16h pa0 = frag_a_h(Pf, 64, mw * 16, 0, lane);
    const v16h pa1 = frag_a_h(Pf, 64, mw * 16, 32, lane);
#pragma unroll
    for (int nt = 0; nt < 16; ++nt) {
      const int n0 = nw * 256 + nt * 16;
      const v16h vb0 = frag_b_kn_h(Vc, 512, 0, n0, lane);
      const v16h vb1 = frag_b_kn_h(Vc, 512, 32, n0, lane);
      acc[nt] = wmma16(pa0, vb0, acc[nt]);
      acc[nt] = wmma16(pa1, vb1, acc[nt]);
    }
  }
  __syncthreads();
  if (tid < 64) {
    const size_t off = ((size_t)b * NH + h) * SEQ + sq0 + tid;
    const float nn = fmaxf(fabsf(rowsum[tid]), __expf(-m_row[off])) + 1e-6f;
    rowE[tid] = 1.f / nn;  // reuse as inverse normalizer
  }
  __syncthreads();
  const int ch = (lane & 16) >> 1;
#pragma unroll
  for (int nt = 0; nt < 16; ++nt) {
    const int col = nw * 256 + nt * 16 + (lane & 15);
#pragma unroll
    for (int r = 0; r < 8; ++r) {
      const int row = mw * 16 + ch + r;
      const int s = sq0 + row;
      h_out[(size_t)(b * SEQ + s) * INNER + h * DH + col] =
          acc[nt][r] * rowE[row];
    }
  }
}

// ---------------------------------------------------------------------------
// MultiHeadLayerNorm over DH + learnable skip from conv branch + SiLU(z) gate.
// Output written in f16 for the down-projection GEMM.
// ---------------------------------------------------------------------------
__global__ __launch_bounds__(128) void ln_skip_gate_kernel(
    const float* __restrict__ h_att, const float* __restrict__ lnw,
    const float* __restrict__ skip, const float* __restrict__ xc,
    const float* __restrict__ x_inner, _Float16* __restrict__ h_state) {
  const int bs = blockIdx.x >> 2;
  const int h = blockIdx.x & 3;
  const int tid = threadIdx.x;
  float vals[4];
  float ssum = 0.f, ssq = 0.f;
#pragma unroll
  for (int r = 0; r < 4; ++r) {
    const int d = tid + r * 128;
    const float vv = h_att[(size_t)bs * INNER + h * DH + d];
    vals[r] = vv;
    ssum += vv;
    ssq += vv * vv;
  }
  __shared__ float red[128];
  red[tid] = ssum;
  __syncthreads();
  for (int o = 64; o > 0; o >>= 1) {
    if (tid < o) red[tid] += red[tid + o];
    __syncthreads();
  }
  const float mu = red[0] * (1.f / DH);
  __syncthreads();
  red[tid] = ssq;
  __syncthreads();
  for (int o = 64; o > 0; o >>= 1) {
    if (tid < o) red[tid] += red[tid + o];
    __syncthreads();
  }
  const float var = red[0] * (1.f / DH) - mu * mu;
  const float rstd = rsqrtf(var + 1e-5f);
#pragma unroll
  for (int r = 0; r < 4; ++r) {
    const int d = tid + r * 128;
    const int c = h * DH + d;
    const float hn = (vals[r] - mu) * rstd * lnw[c];
    const float hs = hn + skip[c] * xc[(size_t)bs * INNER + c];
    const float z = x_inner[(size_t)bs * XLD + INNER + c];
    h_state[(size_t)bs * INNER + c] = (_Float16)(hs * (z / (1.f + __expf(-z))));
  }
}

// ---------------------------------------------------------------------------
extern "C" void kernel_launch(void* const* d_in, const int* in_sizes, int n_in,
                              void* d_out, int out_size, void* d_ws,
                              size_t ws_size, hipStream_t stream) {
  (void)in_sizes; (void)n_in; (void)out_size; (void)ws_size;
  const float* x     = (const float*)d_in[0];
  const float* w_up  = (const float*)d_in[1];
  const float* cw    = (const float*)d_in[2];
  const float* cb    = (const float*)d_in[3];
  const float* wq    = (const float*)d_in[4];
  const float* wk    = (const float*)d_in[5];
  const float* wv    = (const float*)d_in[6];
  const float* wig   = (const float*)d_in[7];
  const float* big   = (const float*)d_in[8];
  const float* wfg   = (const float*)d_in[9];
  const float* bfg   = (const float*)d_in[10];
  const float* lnw   = (const float*)d_in[11];
  const float* skip  = (const float*)d_in[12];
  const float* wdown = (const float*)d_in[13];
  float* out = (float*)d_out;

  char* wsp = (char*)d_ws;
  auto alloc = [&](size_t bytes) -> char* {
    char* p = wsp;
    wsp += (bytes + 255) & ~(size_t)255;
    return p;
  };
  _Float16* xh      = (_Float16*)alloc((size_t)BS_TOT * EMB * 2);
  _Float16* wuph    = (_Float16*)alloc((size_t)2 * INNER * EMB * 2);
  _Float16* wdownh  = (_Float16*)alloc((size_t)EMB * INNER * 2);
  float*    x_inner = (float*)alloc((size_t)BS_TOT * XLD * 4);
  float*    xc      = (float*)alloc((size_t)BS_TOT * INNER * 4);
  _Float16* qh      = (_Float16*)alloc((size_t)BS_TOT * INNER * 2);
  _Float16* kh      = (_Float16*)alloc((size_t)BS_TOT * INNER * 2);
  _Float16* vh      = (_Float16*)alloc((size_t)BS_TOT * INNER * 2);
  float*    igp     = (float*)alloc((size_t)BDIM * NH * SEQ * 4);
  float*    fgp     = (float*)alloc((size_t)BDIM * NH * SEQ * 4);
  float*    a_cum   = (float*)alloc((size_t)BDIM * NH * SEQ * 4);
  float*    g_vec   = (float*)alloc((size_t)BDIM * NH * SEQ * 4);
  float*    m_rowv  = (float*)alloc((size_t)BDIM * NH * SEQ * 4);
  float*    h_att   = (float*)alloc((size_t)BS_TOT * INNER * 4);
  _Float16* h_sth   = (_Float16*)alloc((size_t)BS_TOT * INNER * 2);

  // 0) one-time f32 -> f16 of GEMM operands
  cvt_f32_f16_kernel<<<(BS_TOT * EMB) / 1024, 256, 0, stream>>>(x, xh);
  cvt_f32_f16_kernel<<<(2 * INNER * EMB) / 1024, 256, 0, stream>>>(w_up, wuph);
  cvt_f32_f16_kernel<<<(EMB * INNER) / 1024, 256, 0, stream>>>(wdown, wdownh);
  // 1) up-projection: x_inner = x @ w_up^T   (8192 x 4096, K=1024)
  gemm_h_bt_wmma<<<dim3((2 * INNER) / 128, BS_TOT / 128), 256, 0, stream>>>(
      xh, wuph, x_inner, BS_TOT, 2 * INNER, EMB, EMB, EMB, XLD);
  // 2) causal depthwise conv + SiLU on x_m half
  conv_silu_kernel<<<(BS_TOT * INNER) / 256, 256, 0, stream>>>(x_inner, cw, cb,
                                                               xc);
  // 3) headwise 4x4 q,k,v projections (f16 outputs)
  qkv_kernel<<<(BS_TOT * INNER) / 256, 256, 0, stream>>>(xc, x_inner, wq, wk,
                                                         wv, qh, kh, vh);
  // 4) gate pre-activations
  gate_kernel<<<dim3(BS_TOT, NH), 256, 0, stream>>>(qh, kh, vh, wig, big, wfg,
                                                    bfg, igp, fgp);
  // 5) per-(b,h) prefix scans: a, g, row-max m
  scan_kernel<<<BDIM * NH, 256, 0, stream>>>(fgp, igp, a_cum, g_vec, m_rowv);
  // 6) fused quadratic mLSTM core
  mlstm_attn_kernel<<<dim3(SEQ / 64, NH, BDIM), 256, 0, stream>>>(
      qh, kh, vh, a_cum, g_vec, m_rowv, h_att);
  // 7) per-head layernorm + skip + output gate (f16 output)
  ln_skip_gate_kernel<<<BS_TOT * NH, 128, 0, stream>>>(h_att, lnw, skip, xc,
                                                       x_inner, h_sth);
  // 8) down-projection: out = h_state @ w_down^T  (8192 x 1024, K=2048)
  gemm_h_bt_wmma<<<dim3(EMB / 128, BS_TOT / 128), 256, 0, stream>>>(
      h_sth, wdownh, out, BS_TOT, EMB, INNER, INNER, INNER, EMB);
}
